// OSM_CAA_Loss_84602265797196
// MI455X (gfx1250) — matrix operand order repaired and checked by reading, not compiled, not run on hardware
//
#include <hip/hip_runtime.h>
#include <hip/hip_bf16.h>
#include <math.h>

#define N_ROWS 128
#define DDIM   2048
#define CDIM   1000
#define CPAD   1024

#define ALPHA  1.2f
#define L_MIX  0.5f
#define SIGMA  0.8f
#define EPSF   1e-12f

typedef __attribute__((ext_vector_type(16))) __bf16 v16bf;
typedef __attribute__((ext_vector_type(8)))  float  v8f;

__device__ __forceinline__ unsigned short f32_to_bf16_rne(float f) {
    unsigned int u = __builtin_bit_cast(unsigned int, f);
    u = (u + 0x7FFFu + ((u >> 16) & 1u)) >> 16;
    return (unsigned short)u;
}

// ---------------------------------------------------------------------------
// Kernel 1: L2-normalize each row of x (128 x 2048), emit bf16.
// ---------------------------------------------------------------------------
__global__ void row_normalize_k(const float* __restrict__ x,
                                unsigned short* __restrict__ xnbf) {
    __shared__ float red[256];
    const int i = blockIdx.x;
    const int tid = threadIdx.x;
    float ss = 0.0f;
    for (int k = tid; k < DDIM; k += 256) {
        float v = x[i * DDIM + k];
        ss += v * v;
    }
    red[tid] = ss;
    __syncthreads();
    for (int s = 128; s > 0; s >>= 1) {
        if (tid < s) red[tid] += red[tid + s];
        __syncthreads();
    }
    const float scale = 1.0f / fmaxf(sqrtf(red[0]), EPSF);
    for (int k = tid; k < DDIM; k += 256) {
        xnbf[i * DDIM + k] = f32_to_bf16_rne(x[i * DDIM + k] * scale);
    }
}

// ---------------------------------------------------------------------------
// Kernel 2: transpose embd (2048 x 1000) -> bf16 embdT (1024 x 2048), zero pad.
// ---------------------------------------------------------------------------
__global__ void embd_transpose_k(const float* __restrict__ embd,
                                 unsigned short* __restrict__ ebT) {
    const int idx = blockIdx.x * 256 + threadIdx.x;  // 1024*2048 total
    const int k = idx & (DDIM - 1);
    const int c = idx >> 11;
    float v = (c < CDIM) ? embd[k * CDIM + c] : 0.0f;
    ebT[c * DDIM + k] = f32_to_bf16_rne(v);
}

// ---------------------------------------------------------------------------
// Kernel 3: column norms of embd: colnorm[c] = max(||embd[:,c]||, eps).
// ---------------------------------------------------------------------------
__global__ void col_norm_k(const float* __restrict__ embd,
                           float* __restrict__ colnorm) {
    const int c = blockIdx.x * 256 + threadIdx.x;
    if (c >= CPAD) return;
    float s = 0.0f;
    if (c < CDIM) {
        for (int k = 0; k < DDIM; ++k) {
            float v = embd[k * CDIM + c];
            s += v * v;
        }
    }
    colnorm[c] = fmaxf(sqrtf(s), EPSF);
}

// ---------------------------------------------------------------------------
// Kernel 4: P = xn @ embd via v_wmma_f32_16x16x32_bf16.
//   A = xn bf16 (128 x 2048 row-major), B = embdT bf16 (1024 x 2048, i.e. B^T).
//   Grid: 64 blocks x 256 threads = 512 waves; one 16x16 tile of P per wave.
// ---------------------------------------------------------------------------
__global__ void gemm_p_wmma_k(const unsigned short* __restrict__ xnbf,
                              const unsigned short* __restrict__ ebT,
                              float* __restrict__ P) {
    const int lane = threadIdx.x & 31;
    const int wave = threadIdx.x >> 5;
    const int tile = blockIdx.x * 8 + wave;   // 0..511
    const int tm   = tile & 7;                // row tile (M/16 = 8)
    const int tc   = tile >> 3;               // col tile (CPAD/16 = 64)
    const int half = lane >> 4;               // 0 or 1
    const int l15  = lane & 15;

    const unsigned int* A32 = (const unsigned int*)xnbf;
    const unsigned int* B32 = (const unsigned int*)ebT;

    const int mrow = tm * 16 + l15;           // A-matrix row for this lane
    const int ncol = tc * 16 + l15;           // B-matrix column for this lane

    v8f acc = {};
    for (int k0 = 0; k0 < DDIM; k0 += 32) {
        union { unsigned int u[8]; v16bf v; } Af, Bf;
#pragma unroll
        for (int vi = 0; vi < 8; ++vi) {
            // A 16x32 bf16 layout: VGPR vi holds K = (vi>>2)*16 + half*8 + (vi&3)*2 .. +1
            const int ka = k0 + ((vi >> 2) << 4) + (half << 3) + ((vi & 3) << 1);
            Af.u[vi] = A32[(mrow * DDIM + ka) >> 1];
            // B 32x16 bf16 layout: VGPR vi holds K = (vi>>2)*16 + half*8 + (vi&3)*2 .. +1
            const int kb = k0 + ((vi >> 2) << 4) + (half << 3) + ((vi & 3) << 1);
            Bf.u[vi] = B32[(ncol * DDIM + kb) >> 1];
        }
        acc = __builtin_amdgcn_wmma_f32_16x16x32_bf16(
            /*neg_a=*/false, Af.v, /*neg_b=*/false, Bf.v,
            /*c_mod=*/(short)0, acc, /*reuse_a=*/false, /*reuse_b=*/false);
    }

#pragma unroll
    for (int r = 0; r < 8; ++r) {
        const int m = tm * 16 + half * 8 + r;        // C/D layout: VGPR r -> M = r + half*8
        P[m * CPAD + tc * 16 + l15] = acc[r];
    }
}

// ---------------------------------------------------------------------------
// Kernel 5: atten[i] = exp(P[i,lab]/||e_lab||) / sum_c exp(P[i,c]/||e_c||).
// ---------------------------------------------------------------------------
__global__ void atten_k(const float* __restrict__ P,
                        const float* __restrict__ colnorm,
                        const int* __restrict__ labels,
                        float* __restrict__ atten) {
    __shared__ float rsum[256];
    __shared__ float rnum[256];
    const int i = blockIdx.x;
    const int tid = threadIdx.x;
    const int lab = labels[i];
    float s = 0.0f, num = 0.0f;
    for (int c = tid; c < CDIM; c += 256) {
        float v = expf(P[i * CPAD + c] / colnorm[c]);
        s += v;
        if (c == lab) num = v;
    }
    rsum[tid] = s;
    rnum[tid] = num;
    __syncthreads();
    for (int st = 128; st > 0; st >>= 1) {
        if (tid < st) { rsum[tid] += rsum[tid + st]; rnum[tid] += rnum[tid + st]; }
        __syncthreads();
    }
    if (tid == 0) atten[i] = rnum[0] / rsum[0];
}

// ---------------------------------------------------------------------------
// Kernel 6: pairwise reduction -> scalar loss.
//   s = M.sum(0) == 0 analytically (row-sum antisymmetry), so
//   dist = clip(M@s + EPS, EPS) = EPS (constant).
// ---------------------------------------------------------------------------
__global__ void final_loss_k(const int* __restrict__ labels,
                             const float* __restrict__ atten,
                             float* __restrict__ out) {
    __shared__ float rwp[256], rwpd[256], rwn[256], rwns[256];
    const int tid = threadIdx.x;

    const float dist  = EPSF;
    const float dist2 = dist * dist;
    const float Sexp  = expf(-dist2 / (SIGMA * SIGMA));
    const float Sneg  = fmaxf(ALPHA - dist, EPSF);

    float wp = 0.0f, wpd = 0.0f, wn = 0.0f, wns = 0.0f;
    for (int idx = tid; idx < N_ROWS * N_ROWS; idx += 256) {
        const int i = idx >> 7;
        const int j = idx & (N_ROWS - 1);
        if (i == j) continue;
        const float a = fminf(atten[i], atten[j]);
        if (labels[i] == labels[j]) {
            const float w = Sexp * a;
            wp  += w;
            wpd += w * dist2;
        } else {
            const float w = Sneg * a;
            wn  += w;
            wns += w * Sneg * Sneg;
        }
    }
    rwp[tid] = wp; rwpd[tid] = wpd; rwn[tid] = wn; rwns[tid] = wns;
    __syncthreads();
    for (int st = 128; st > 0; st >>= 1) {
        if (tid < st) {
            rwp[tid]  += rwp[tid + st];
            rwpd[tid] += rwpd[tid + st];
            rwn[tid]  += rwn[tid + st];
            rwns[tid] += rwns[tid + st];
        }
        __syncthreads();
    }
    if (tid == 0) {
        const float LP = 0.5f * rwpd[0] / rwp[0];
        const float LN = 0.5f * rwns[0] / rwn[0];
        out[0] = (1.0f - L_MIX) * LP + L_MIX * LN;
    }
}

// ---------------------------------------------------------------------------
// Host launcher
// ---------------------------------------------------------------------------
extern "C" void kernel_launch(void* const* d_in, const int* in_sizes, int n_in,
                              void* d_out, int out_size, void* d_ws, size_t ws_size,
                              hipStream_t stream) {
    (void)in_sizes; (void)n_in; (void)out_size; (void)ws_size;
    const float* x      = (const float*)d_in[0];
    const int*   labels = (const int*)d_in[1];
    const float* embd   = (const float*)d_in[2];
    float* out = (float*)d_out;

    char* ws = (char*)d_ws;
    unsigned short* xnbf    = (unsigned short*)(ws);                       // 128*2048*2   = 512 KB
    unsigned short* ebT     = (unsigned short*)(ws + 524288);              // 1024*2048*2  = 4 MB
    float*          P       = (float*)(ws + 524288 + 4194304);             // 128*1024*4   = 512 KB
    float*          colnorm = (float*)(ws + 524288 + 4194304 + 524288);    // 1024*4
    float*          atten   = (float*)(ws + 524288 + 4194304 + 524288 + 4096);

    row_normalize_k <<<N_ROWS, 256, 0, stream>>>(x, xnbf);
    embd_transpose_k<<<(CPAD * DDIM) / 256, 256, 0, stream>>>(embd, ebT);
    col_norm_k      <<<CPAD / 256, 256, 0, stream>>>(embd, colnorm);
    gemm_p_wmma_k   <<<64, 256, 0, stream>>>(xnbf, ebT, P);
    atten_k         <<<N_ROWS, 256, 0, stream>>>(P, colnorm, labels, atten);
    final_loss_k    <<<1, 256, 0, stream>>>(labels, atten, out);
}
